// ConvBranch_3633542332580
// MI455X (gfx1250) — matrix operand order repaired and checked by reading, not compiled, not run on hardware
//
#include <hip/hip_runtime.h>
#include <hip/hip_bf16.h>
#include <math.h>
#include <stdint.h>

// ---------------------------------------------------------------- constants
#define B_N      4
#define T_N      8192
#define DIN_     256
#define DMODEL_  512
#define NLAYERS_ 4
#define STRIDE_  4
#define KTAPS_   8
#define DSTATE_  16
#define DCONV_   4
#define DINNER_  1024
#define DTRANK_  32
#define L_N      2048                 // T / STRIDE
#define M_ROWS   (B_N * L_N)         // 8192 token rows

typedef __bf16 bf16_t;
typedef __attribute__((ext_vector_type(16))) __bf16 bfx16;
typedef __attribute__((ext_vector_type(8)))  __bf16 bfx8;
typedef __attribute__((ext_vector_type(8)))  float  fx8;
typedef unsigned int u32x4 __attribute__((ext_vector_type(4)));
typedef int          i32x4 __attribute__((ext_vector_type(4)));
typedef int          i32x8 __attribute__((ext_vector_type(8)));

__device__ __forceinline__ unsigned short f32_to_bf16_bits(float f) {
  union { __bf16 b; unsigned short u; } cv;
  cv.b = (__bf16)f;
  return cv.u;
}

// ---------------------------------------------------------------- fp32 -> bf16
__global__ void __launch_bounds__(256) cvt_bf16_kernel(
    const float* __restrict__ in, unsigned short* __restrict__ out, int n) {
  int i = blockIdx.x * 256 + threadIdx.x;
  if (i < n) out[i] = f32_to_bf16_bits(in[i]);
}

// ---------------------------------------------------------------- stem: strided conv + GELU
__global__ void __launch_bounds__(256) stem_kernel(
    const float* __restrict__ x, const float* __restrict__ w,
    const float* __restrict__ bias, float* __restrict__ h) {
  const int m = blockIdx.x;               // 0..8191 = (b, l)
  const int b = m >> 11, l = m & (L_N - 1);
  __shared__ float sx[KTAPS_ * DIN_];     // 8 taps x 256 ch
  for (int id = threadIdx.x; id < KTAPS_ * DIN_; id += 256) {
    const int k = id >> 8, c = id & (DIN_ - 1);
    const int t = l * STRIDE_ - (KTAPS_ - 1) + k;
    sx[id] = (t >= 0) ? x[((size_t)b * T_N + t) * DIN_ + c] : 0.f;
  }
  __syncthreads();
  for (int o = threadIdx.x; o < DMODEL_; o += 256) {
    float acc = bias[o];
    const float* wr = w + (size_t)o * (DIN_ * KTAPS_);   // (o, c, k)
    for (int c = 0; c < DIN_; ++c)
      for (int k = 0; k < KTAPS_; ++k)
        acc = fmaf(sx[k * DIN_ + c], wr[c * KTAPS_ + k], acc);
    const float g = 0.5f * acc * (1.f + erff(acc * 0.70710678118654752f));
    h[(size_t)m * DMODEL_ + o] = g;
  }
}

// ---------------------------------------------------------------- layernorm -> bf16 (one wave per row)
__global__ void __launch_bounds__(256) ln_to_bf16_kernel(
    const float* __restrict__ h, const float* __restrict__ w,
    const float* __restrict__ bias, unsigned short* __restrict__ out) {
  const int lane = threadIdx.x & 31;
  const int row  = blockIdx.x * 8 + (threadIdx.x >> 5);
  const float* hr = h + (size_t)row * DMODEL_;
  float vals[16];
  float s = 0.f;
  for (int i = 0; i < 16; ++i) { vals[i] = hr[lane + 32 * i]; s += vals[i]; }
  for (int off = 16; off > 0; off >>= 1) s += __shfl_xor(s, off, 32);
  const float mean = s * (1.f / (float)DMODEL_);
  float v = 0.f;
  for (int i = 0; i < 16; ++i) { const float d = vals[i] - mean; v = fmaf(d, d, v); }
  for (int off = 16; off > 0; off >>= 1) v += __shfl_xor(v, off, 32);
  const float inv = rsqrtf(v * (1.f / (float)DMODEL_) + 1e-5f);
  for (int i = 0; i < 16; ++i) {
    const int c = lane + 32 * i;
    out[(size_t)row * DMODEL_ + c] = f32_to_bf16_bits((vals[i] - mean) * inv * w[c] + bias[c]);
  }
}

// ---------------------------------------------------------------- final LN + repeat x4 into output
__global__ void __launch_bounds__(256) final_ln_repeat_kernel(
    const float* __restrict__ h, const float* __restrict__ w,
    const float* __restrict__ bias, float* __restrict__ out) {
  const int lane = threadIdx.x & 31;
  const int row  = blockIdx.x * 8 + (threadIdx.x >> 5);
  const float* hr = h + (size_t)row * DMODEL_;
  float vals[16];
  float s = 0.f;
  for (int i = 0; i < 16; ++i) { vals[i] = hr[lane + 32 * i]; s += vals[i]; }
  for (int off = 16; off > 0; off >>= 1) s += __shfl_xor(s, off, 32);
  const float mean = s * (1.f / (float)DMODEL_);
  float v = 0.f;
  for (int i = 0; i < 16; ++i) { const float d = vals[i] - mean; v = fmaf(d, d, v); }
  for (int off = 16; off > 0; off >>= 1) v += __shfl_xor(v, off, 32);
  const float inv = rsqrtf(v * (1.f / (float)DMODEL_) + 1e-5f);
  float* o0 = out + (size_t)row * STRIDE_ * DMODEL_;
  for (int i = 0; i < 16; ++i) {
    const int c = lane + 32 * i;
    const float o = (vals[i] - mean) * inv * w[c] + bias[c];
    o0[c] = o; o0[DMODEL_ + c] = o; o0[2 * DMODEL_ + c] = o; o0[3 * DMODEL_ + c] = o;
  }
}

// ---------------------------------------------------------------- TDM tile DMA (gfx1250)
#define GBLK 64
#define LDSK 40   // padded LDS row stride (bf16 elems): 80B (TDM pad: +4 DWORDs / 16 DWORDs)

#if defined(__HIP_DEVICE_COMPILE__) && __has_builtin(__builtin_amdgcn_tensor_load_to_lds)
#define USE_TDM 1
#else
#define USE_TDM 0
#endif

#if USE_TDM
// Load a (rows x 32) bf16 tile from a row-major (tensor_rows x klen) matrix
// into LDS with an 80-byte row pitch, via one TENSOR_LOAD_TO_LDS (D# per ISA 8.3/8.4).
__device__ __forceinline__ void tdm_load_tile_2d(
    const bf16_t* gsrc, unsigned lds_off, int klen, int rows, int tensor_rows) {
  const unsigned long long ga = (unsigned long long)(uintptr_t)gsrc;
  u32x4 g0;
  g0.x = 1u;                                                    // count=1 (valid user D#)
  g0.y = lds_off;                                               // lds_addr (bytes)
  g0.z = (unsigned)ga;                                          // global_addr[31:0]
  g0.w = (unsigned)((ga >> 32) & 0x01ffffffull) | (2u << 30);   // global_addr[56:32] | type=2
  i32x8 g1;
  g1[0] = (1 << 16)      // data_size = 2 bytes (bf16)
        | (1 << 20)      // pad_enable
        | (3 << 22)      // pad_interval: 16 DWORDs (one 64B tile row)
        | (3 << 25);     // pad_amount:   4 DWORDs (16B -> 80B row pitch)
  g1[1] = (klen & 0xffff) << 16;                                // tensor_dim0[15:0]
  g1[2] = (((unsigned)klen >> 16) & 0xffff)                     // tensor_dim0[31:16]
        | ((tensor_rows & 0xffff) << 16);                       // tensor_dim1[15:0]
  g1[3] = (((unsigned)tensor_rows >> 16) & 0xffff)              // tensor_dim1[31:16]
        | (32 << 16);                                           // tile_dim0 = 32 elems
  g1[4] = rows & 0xffff;                                        // tile_dim1 (tile_dim2 = 0)
  g1[5] = klen;                                                 // tensor_dim0_stride[31:0]
  g1[6] = 0;                                                    // stride hi / dim1_stride lo
  g1[7] = 0;
  const i32x4 z4 = {0, 0, 0, 0};
#if __clang_major__ >= 23
  const i32x8 z8 = {0, 0, 0, 0, 0, 0, 0, 0};
  __builtin_amdgcn_tensor_load_to_lds(g0, g1, z4, z4, z8, 0);
#else
  __builtin_amdgcn_tensor_load_to_lds(g0, g1, z4, z4, 0);
#endif
}

__device__ __forceinline__ void tdm_wait0() {
#if __has_builtin(__builtin_amdgcn_s_wait_tensorcnt)
  __builtin_amdgcn_s_wait_tensorcnt(0);
#else
  asm volatile("s_wait_tensorcnt 0x0" ::: "memory");
#endif
}
#endif // USE_TDM

// ---------------------------------------------------------------- bf16 WMMA GEMM
// C[M,N] (f32, + compile-time optional residual) = A[M,K] (bf16 rm) * W[N,K]^T (bf16 rm)
// 128 threads = 4 waves; block tile 64x64; wave tile 32x32 = 2x2 WMMA.
// TDM double-buffered LDS: wave 0 DMAs tile kb+1 while all waves run WMMA on tile kb.
template <bool ADD>
__global__ void __launch_bounds__(128) gemm_bf16_wmma_kernel(
    const unsigned short* __restrict__ Abits,
    const unsigned short* __restrict__ Wbits,
    float* __restrict__ C, const float* __restrict__ addC,
    int M, int N, int K) {
  const bf16_t* A = reinterpret_cast<const bf16_t*>(Abits);
  const bf16_t* W = reinterpret_cast<const bf16_t*>(Wbits);

  __shared__ __align__(16) bf16_t As[2][GBLK * LDSK];
  __shared__ __align__(16) bf16_t Bs[2][GBLK * LDSK];

  const int tid  = threadIdx.x;
  const int lane = tid & 31;
  const int wave = tid >> 5;
  const int wm = (wave & 1) * 32;
  const int wn = (wave >> 1) * 32;
  const int m0 = blockIdx.y * GBLK;
  const int n0 = blockIdx.x * GBLK;

  const fx8 zacc = {0.f, 0.f, 0.f, 0.f, 0.f, 0.f, 0.f, 0.f};
  fx8 acc[2][2];
  acc[0][0] = zacc; acc[0][1] = zacc; acc[1][0] = zacc; acc[1][1] = zacc;

  const int ksteps = K >> 5;

#if USE_TDM
  if (wave == 0) {
    tdm_load_tile_2d(A + (size_t)m0 * K, (unsigned)(uintptr_t)&As[0][0], K, GBLK, M);
    tdm_load_tile_2d(W + (size_t)n0 * K, (unsigned)(uintptr_t)&Bs[0][0], K, GBLK, N);
    tdm_wait0();
  }
  __syncthreads();

  for (int kb = 0; kb < ksteps; ++kb) {
    const int cur = kb & 1;
    const bool more = (kb + 1) < ksteps;
    if (wave == 0 && more) {                      // DMA next tile into other buffer
      const int k1 = (kb + 1) << 5;
      tdm_load_tile_2d(A + (size_t)m0 * K + k1, (unsigned)(uintptr_t)&As[cur ^ 1][0], K, GBLK, M);
      tdm_load_tile_2d(W + (size_t)n0 * K + k1, (unsigned)(uintptr_t)&Bs[cur ^ 1][0], K, GBLK, N);
    }

    const int lrow  = lane & 15;
    const int lhalf = lane >> 4;
    bfx16 afrag[2], bfrag[2];
    for (int i = 0; i < 2; ++i) {
      const bf16_t* ap = &As[cur][(wm + i * 16 + lrow) * LDSK + lhalf * 8];
      reinterpret_cast<bfx8*>(&afrag[i])[0] = *reinterpret_cast<const bfx8*>(ap);
      reinterpret_cast<bfx8*>(&afrag[i])[1] = *reinterpret_cast<const bfx8*>(ap + 16);
    }
    for (int j = 0; j < 2; ++j) {
      const bf16_t* bp = &Bs[cur][(wn + j * 16 + lrow) * LDSK + lhalf * 8];
      reinterpret_cast<bfx8*>(&bfrag[j])[0] = *reinterpret_cast<const bfx8*>(bp);
      reinterpret_cast<bfx8*>(&bfrag[j])[1] = *reinterpret_cast<const bfx8*>(bp + 16);
    }
    for (int i = 0; i < 2; ++i)
      for (int j = 0; j < 2; ++j)
        acc[i][j] = __builtin_amdgcn_wmma_f32_16x16x32_bf16(
            false, afrag[i], false, bfrag[j], (short)0, acc[i][j], false, false);

    if (wave == 0 && more) tdm_wait0();           // next tile landed
    __syncthreads();                              // publish to all waves
  }
#else
  // Fallback: synchronous global->LDS staging (single buffer, two barriers/step)
  for (int kb = 0; kb < ksteps; ++kb) {
    const int k0 = kb << 5;
    for (int id = tid; id < GBLK * 4; id += 128) {
      const int row = id >> 2;
      const int c8  = (id & 3) << 3;
      *reinterpret_cast<uint4*>(&As[0][row * LDSK + c8]) =
          *reinterpret_cast<const uint4*>(&A[(size_t)(m0 + row) * K + k0 + c8]);
      *reinterpret_cast<uint4*>(&Bs[0][row * LDSK + c8]) =
          *reinterpret_cast<const uint4*>(&W[(size_t)(n0 + row) * K + k0 + c8]);
    }
    if (kb + 1 < ksteps) {
      const int row = tid & (GBLK - 1);
      __builtin_prefetch(&A[(size_t)(m0 + row) * K + k0 + 32], 0, 1);
      __builtin_prefetch(&W[(size_t)(n0 + row) * K + k0 + 32], 0, 1);
    }
    __syncthreads();
    const int lrow  = lane & 15;
    const int lhalf = lane >> 4;
    bfx16 afrag[2], bfrag[2];
    for (int i = 0; i < 2; ++i) {
      const bf16_t* ap = &As[0][(wm + i * 16 + lrow) * LDSK + lhalf * 8];
      reinterpret_cast<bfx8*>(&afrag[i])[0] = *reinterpret_cast<const bfx8*>(ap);
      reinterpret_cast<bfx8*>(&afrag[i])[1] = *reinterpret_cast<const bfx8*>(ap + 16);
    }
    for (int j = 0; j < 2; ++j) {
      const bf16_t* bp = &Bs[0][(wn + j * 16 + lrow) * LDSK + lhalf * 8];
      reinterpret_cast<bfx8*>(&bfrag[j])[0] = *reinterpret_cast<const bfx8*>(bp);
      reinterpret_cast<bfx8*>(&bfrag[j])[1] = *reinterpret_cast<const bfx8*>(bp + 16);
    }
    for (int i = 0; i < 2; ++i)
      for (int j = 0; j < 2; ++j)
        acc[i][j] = __builtin_amdgcn_wmma_f32_16x16x32_bf16(
            false, afrag[i], false, bfrag[j], (short)0, acc[i][j], false, false);
    __syncthreads();
  }
#endif

  // epilogue: branch-free; residual loads batched up front when ADD
  const int lrow  = lane & 15;
  const int lhalf = lane >> 4;
  float res[2][2][8];
  if (ADD) {
    for (int i = 0; i < 2; ++i)
      for (int j = 0; j < 2; ++j) {
        const size_t base = (size_t)(m0 + wm + i * 16 + lhalf * 8) * N + (n0 + wn + j * 16 + lrow);
        for (int r = 0; r < 8; ++r)
          res[i][j][r] = addC[base + (size_t)r * N];
      }
  }
  for (int i = 0; i < 2; ++i)
    for (int j = 0; j < 2; ++j) {
      const size_t base = (size_t)(m0 + wm + i * 16 + lhalf * 8) * N + (n0 + wn + j * 16 + lrow);
      for (int r = 0; r < 8; ++r) {
        float v = acc[i][j][r];
        if (ADD) v += res[i][j][r];
        C[base + (size_t)r * N] = v;
      }
    }
}

// ---------------------------------------------------------------- depthwise causal conv (K=4) + SiLU
__global__ void __launch_bounds__(256) dwconv_silu_kernel(
    const float* __restrict__ xz, const float* __restrict__ w,
    const float* __restrict__ bias, float* __restrict__ u,
    unsigned short* __restrict__ xb) {
  const int idx = blockIdx.x * 256 + threadIdx.x;      // over M_ROWS * DINNER
  const int d = idx & (DINNER_ - 1);
  const int m = idx >> 10;
  const int b = m >> 11, l = m & (L_N - 1);
  float acc = bias[d];
  const float* wd = w + d * DCONV_;
  for (int j = 0; j < DCONV_; ++j) {
    const int lj = l - (DCONV_ - 1) + j;
    if (lj >= 0)
      acc = fmaf(xz[((size_t)(b * L_N + lj)) * (2 * DINNER_) + d], wd[j], acc);
  }
  const float s = acc / (1.f + __expf(-acc));          // silu
  u[(size_t)m * DINNER_ + d] = s;
  xb[(size_t)m * DINNER_ + d] = f32_to_bf16_bits(s);
}

// ---------------------------------------------------------------- dt = softplus(dbc[:,:32] @ dtw^T + dtb)
__global__ void __launch_bounds__(256) dt_softplus_kernel(
    const float* __restrict__ dbc, const float* __restrict__ dtw,
    const float* __restrict__ dtb, float* __restrict__ dt) {
  const int m = blockIdx.y;
  const int d = blockIdx.x * 256 + threadIdx.x;
  __shared__ float r[DTRANK_];
  if (threadIdx.x < DTRANK_) r[threadIdx.x] = dbc[(size_t)m * 64 + threadIdx.x];
  __syncthreads();
  float acc = dtb[d];
  const float* wr = dtw + (size_t)d * DTRANK_;
  for (int k = 0; k < DTRANK_; ++k) acc = fmaf(r[k], wr[k], acc);
  const float sp = (acc > 20.f) ? acc : log1pf(__expf(acc));
  dt[(size_t)m * DINNER_ + d] = sp;
}

// ---------------------------------------------------------------- selective scan + gate -> y (bf16)
__global__ void __launch_bounds__(256) scan_kernel(
    const float* __restrict__ u, const float* __restrict__ dt,
    const float* __restrict__ dbc, const float* __restrict__ xz,
    const float* __restrict__ A_log, const float* __restrict__ Dp,
    unsigned short* __restrict__ y_out) {
  const int b = blockIdx.x;                   // 0..3
  const int d = blockIdx.y * 256 + threadIdx.x;
  float aval[DSTATE_];
  for (int n = 0; n < DSTATE_; ++n) aval[n] = -__expf(A_log[(size_t)d * DSTATE_ + n]);
  const float Dd = Dp[d];
  float hst[DSTATE_];
  for (int n = 0; n < DSTATE_; ++n) hst[n] = 0.f;
  __shared__ float sBC[2 * DSTATE_];          // [0:16)=B_t, [16:32)=C_t
  for (int t = 0; t < L_N; ++t) {
    const size_t row = (size_t)b * L_N + t;
    if (threadIdx.x < 2 * DSTATE_)
      sBC[threadIdx.x] = dbc[row * 64 + DTRANK_ + threadIdx.x];
    __syncthreads();
    const float dtt = dt[row * DINNER_ + d];
    const float ut  = u[row * DINNER_ + d];
    const float du  = dtt * ut;
    float y = 0.f;
    for (int n = 0; n < DSTATE_; ++n) {
      const float dA = __expf(dtt * aval[n]);
      hst[n] = fmaf(hst[n], dA, du * sBC[n]);
      y = fmaf(hst[n], sBC[DSTATE_ + n], y);
    }
    const float z = xz[row * (2 * DINNER_) + DINNER_ + d];
    const float g = z / (1.f + __expf(-z));   // silu(z)
    y_out[row * DINNER_ + d] = f32_to_bf16_bits((y + ut * Dd) * g);
    __syncthreads();
  }
}

// ---------------------------------------------------------------- host orchestration
extern "C" void kernel_launch(void* const* d_in, const int* in_sizes, int n_in,
                              void* d_out, int out_size, void* d_ws, size_t ws_size,
                              hipStream_t stream) {
  (void)in_sizes; (void)n_in; (void)out_size; (void)ws_size;
  const float* x         = (const float*)d_in[0];
  const float* conv_w    = (const float*)d_in[1];
  const float* conv_b    = (const float*)d_in[2];
  const float* in_proj_w = (const float*)d_in[3];
  const float* dconv_w   = (const float*)d_in[4];
  const float* dconv_b   = (const float*)d_in[5];
  const float* x_proj_w  = (const float*)d_in[6];
  const float* dt_proj_w = (const float*)d_in[7];
  const float* dt_proj_b = (const float*)d_in[8];
  const float* A_log     = (const float*)d_in[9];
  const float* D_param   = (const float*)d_in[10];
  const float* out_proj_w= (const float*)d_in[11];
  const float* ln_w      = (const float*)d_in[12];
  const float* ln_b      = (const float*)d_in[13];
  const float* fn_w      = (const float*)d_in[14];
  const float* fn_b      = (const float*)d_in[15];
  float* out = (float*)d_out;

  char* ws = (char*)d_ws;
  size_t off = 0;
  auto take = [&](size_t bytes) -> char* {
    char* p = ws + off;
    off = (off + bytes + 255) & ~(size_t)255;
    return p;
  };
  float*          h_buf   = (float*)take((size_t)M_ROWS * DMODEL_ * 4);
  unsigned short* hn_bf   = (unsigned short*)take((size_t)M_ROWS * DMODEL_ * 2);
  float*          xz_buf  = (float*)take((size_t)M_ROWS * 2 * DINNER_ * 4);
  float*          u_buf   = (float*)take((size_t)M_ROWS * DINNER_ * 4);
  unsigned short* x_bf    = (unsigned short*)take((size_t)M_ROWS * DINNER_ * 2);
  float*          dbc_buf = (float*)take((size_t)M_ROWS * 64 * 4);
  float*          dt_buf  = (float*)take((size_t)M_ROWS * DINNER_ * 4);
  unsigned short* y_bf    = (unsigned short*)take((size_t)M_ROWS * DINNER_ * 2);
  unsigned short* w_in_bf = (unsigned short*)take((size_t)NLAYERS_ * 2 * DINNER_ * DMODEL_ * 2);
  unsigned short* w_xp_bf = (unsigned short*)take((size_t)NLAYERS_ * 64 * DINNER_ * 2);
  unsigned short* w_out_bf= (unsigned short*)take((size_t)NLAYERS_ * DMODEL_ * DINNER_ * 2);

  {
    const int n1 = NLAYERS_ * 2 * DINNER_ * DMODEL_;
    const int n2 = NLAYERS_ * 64 * DINNER_;
    const int n3 = NLAYERS_ * DMODEL_ * DINNER_;
    cvt_bf16_kernel<<<(n1 + 255) / 256, 256, 0, stream>>>(in_proj_w,  w_in_bf,  n1);
    cvt_bf16_kernel<<<(n2 + 255) / 256, 256, 0, stream>>>(x_proj_w,   w_xp_bf,  n2);
    cvt_bf16_kernel<<<(n3 + 255) / 256, 256, 0, stream>>>(out_proj_w, w_out_bf, n3);
  }

  stem_kernel<<<M_ROWS, 256, 0, stream>>>(x, conv_w, conv_b, h_buf);

  for (int i = 0; i < NLAYERS_; ++i) {
    ln_to_bf16_kernel<<<M_ROWS / 8, 256, 0, stream>>>(
        h_buf, ln_w + i * DMODEL_, ln_b + i * DMODEL_, hn_bf);
    // in_proj: (8192x512) x (2048x512)^T -> xz
    gemm_bf16_wmma_kernel<false><<<dim3((2 * DINNER_) / GBLK, M_ROWS / GBLK), 128, 0, stream>>>(
        hn_bf, w_in_bf + (size_t)i * 2 * DINNER_ * DMODEL_, xz_buf, nullptr,
        M_ROWS, 2 * DINNER_, DMODEL_);
    dwconv_silu_kernel<<<(M_ROWS * DINNER_) / 256, 256, 0, stream>>>(
        xz_buf, dconv_w + (size_t)i * DINNER_ * DCONV_, dconv_b + i * DINNER_,
        u_buf, x_bf);
    // x_proj: (8192x1024) x (64x1024)^T -> dbc
    gemm_bf16_wmma_kernel<false><<<dim3(64 / GBLK, M_ROWS / GBLK), 128, 0, stream>>>(
        x_bf, w_xp_bf + (size_t)i * 64 * DINNER_, dbc_buf, nullptr,
        M_ROWS, 64, DINNER_);
    dt_softplus_kernel<<<dim3(DINNER_ / 256, M_ROWS), 256, 0, stream>>>(
        dbc_buf, dt_proj_w + (size_t)i * DINNER_ * DTRANK_, dt_proj_b + i * DINNER_, dt_buf);
    scan_kernel<<<dim3(B_N, DINNER_ / 256), 256, 0, stream>>>(
        u_buf, dt_buf, dbc_buf, xz_buf,
        A_log + (size_t)i * DINNER_ * DSTATE_, D_param + i * DINNER_, y_bf);
    // out_proj with fused residual: h = h + y @ out_w^T
    gemm_bf16_wmma_kernel<true><<<dim3(DMODEL_ / GBLK, M_ROWS / GBLK), 128, 0, stream>>>(
        y_bf, w_out_bf + (size_t)i * DMODEL_ * DINNER_, h_buf, h_buf,
        M_ROWS, DMODEL_, DINNER_);
  }

  final_ln_repeat_kernel<<<M_ROWS / 8, 256, 0, stream>>>(h_buf, fn_w, fn_b, out);
}
